// MultiHeadAttention_33148557591136
// MI455X (gfx1250) — matrix operand order repaired and checked by reading, not compiled, not run on hardware
//
#include <hip/hip_runtime.h>
#include <hip/hip_bf16.h>
#include <cstdint>

typedef __bf16 bf16_t;
typedef __attribute__((ext_vector_type(8)))  __bf16 v8bf;
typedef __attribute__((ext_vector_type(16))) __bf16 v16bf;
typedef __attribute__((ext_vector_type(8)))  float   v8f;

union FragU { v16bf v; v8bf h[2]; };
union AccU  { v8f   v; float f[8]; };

// ---------------------------------------------------------------- convert
__global__ void cvt_f32_to_bf16(const float* __restrict__ s,
                                bf16_t* __restrict__ d, int n) {
  int i = blockIdx.x * blockDim.x + threadIdx.x;
  if (i < n) d[i] = (bf16_t)s[i];
}

// ---------------------------------------------------------------- GEMM
// C[M,N] = A[M,K] * Bw[K,N], bf16 in, f32 accum, double-buffered LDS with
// async global->LDS staging for the A tile (ASYNCcnt pipeline).
// mode 0: N==3072 QKV epilogue -> Q[B,H,T,64], K[B,H,T,64], Vt[B,H,64,T] (bf16)
// mode 1: epilogue Out[row*N+col] = acc + bias[col] (f32)
#define BM 128
#define BN 128
#define BK 32
#define LPAD 40

__global__ __launch_bounds__(256)
void gemm_bf16_wmma(const bf16_t* __restrict__ A, const bf16_t* __restrict__ Bw,
                    int M, int N, int K, int mode,
                    bf16_t* __restrict__ Qb, bf16_t* __restrict__ Kb,
                    bf16_t* __restrict__ Vt,
                    const float* __restrict__ bias, float* __restrict__ Out) {
  __shared__ __align__(16) bf16_t As [2][BM][LPAD];   // [buf][m][k]
  __shared__ __align__(16) bf16_t Bst[2][BN][LPAD];   // [buf][n][k] (transposed)

  const int tid  = threadIdx.x;
  const int lane = tid & 31;
  const int wv   = tid >> 5;       // 0..7
  const int wm   = wv & 1;         // 2 waves along M
  const int wn   = wv >> 1;        // 4 waves along N
  const int m0   = blockIdx.y * BM;
  const int n0   = blockIdx.x * BN;
  const int hlf  = lane >> 4;      // 0 or 1
  const int l16  = lane & 15;
  const int kgA  = hlf * 16;       // A-fragment K base
  const int kgB  = hlf * 16;       // B-fragment K base

  // ---- async A-tile stage: per-lane 16B global -> LDS, tracked by ASYNCcnt
  auto loadA_async = [&](int kb, int buf) {
    #pragma unroll
    for (int t = 0; t < 2; ++t) {
      int v   = t * 256 + tid;
      int row = v >> 2;
      int c8  = (v & 3) * 8;
      unsigned ldsa = (unsigned)(size_t)&As[buf][row][c8];
      unsigned long long ga =
          (unsigned long long)&A[(size_t)(m0 + row) * K + kb + c8];
      asm volatile("global_load_async_to_lds_b128 %0, %1, off"
                   :: "v"(ldsa), "v"(ga) : "memory");
    }
  };
  // ---- B-tile: issue global loads early, commit transpose to LDS later
  uint4 breg[2];
  auto loadB_issue = [&](int kb) {
    #pragma unroll
    for (int t = 0; t < 2; ++t) {
      int v    = t * 256 + tid;
      int krow = v >> 4;
      int c8   = (v & 15) * 8;
      breg[t] = *(const uint4*)&Bw[(size_t)(kb + krow) * N + n0 + c8];
    }
  };
  auto loadB_commit = [&](int buf) {
    #pragma unroll
    for (int t = 0; t < 2; ++t) {
      int v    = t * 256 + tid;
      int krow = v >> 4;
      int c8   = (v & 15) * 8;
      union { uint4 u; bf16_t e[8]; } tmp;
      tmp.u = breg[t];
      #pragma unroll
      for (int j = 0; j < 8; ++j) Bst[buf][c8 + j][krow] = tmp.e[j];
    }
  };

  const v8f zero8 = {0.f,0.f,0.f,0.f,0.f,0.f,0.f,0.f};
  v8f acc[4][2];
  #pragma unroll
  for (int i = 0; i < 4; ++i)
    #pragma unroll
    for (int j = 0; j < 2; ++j) acc[i][j] = zero8;

  // prologue: stage tile 0
  loadA_async(0, 0);
  loadB_issue(0);
  loadB_commit(0);
  asm volatile("s_wait_asynccnt 0x0" ::: "memory");
  __syncthreads();

  const int nk = K / BK;
  for (int ik = 0; ik < nk; ++ik) {
    const int buf  = ik & 1;
    const int nbuf = buf ^ 1;
    if (ik + 1 < nk) {                       // prefetch next tile
      loadA_async((ik + 1) * BK, nbuf);
      loadB_issue((ik + 1) * BK);
    }

    // ---- compute current tile
    FragU bf[2];
    #pragma unroll
    for (int jn = 0; jn < 2; ++jn) {
      int n = wn * 32 + jn * 16 + l16;
      bf[jn].h[0] = *(const v8bf*)&Bst[buf][n][kgB];
      bf[jn].h[1] = *(const v8bf*)&Bst[buf][n][kgB + 8];
    }
    #pragma unroll
    for (int im = 0; im < 4; ++im) {
      int m = wm * 64 + im * 16 + l16;
      FragU af;
      af.h[0] = *(const v8bf*)&As[buf][m][kgA];
      af.h[1] = *(const v8bf*)&As[buf][m][kgA + 8];
      #pragma unroll
      for (int jn = 0; jn < 2; ++jn) {
        acc[im][jn] = __builtin_amdgcn_wmma_f32_16x16x32_bf16(
            false, af.v, false, bf[jn].v, (short)0, acc[im][jn], false, false);
      }
    }

    if (ik + 1 < nk) loadB_commit(nbuf);
    asm volatile("s_wait_asynccnt 0x0" ::: "memory");
    __syncthreads();
  }

  // ---- epilogue
  #pragma unroll
  for (int im = 0; im < 4; ++im) {
    #pragma unroll
    for (int jn = 0; jn < 2; ++jn) {
      AccU u; u.v = acc[im][jn];
      #pragma unroll
      for (int r = 0; r < 8; ++r) {
        int row = m0 + wm * 64 + im * 16 + r + hlf * 8;
        int col = n0 + wn * 32 + jn * 16 + l16;
        float val = u.f[r];
        if (mode == 0) {
          int sec = col >> 10;          // 0=q 1=k 2=v
          int c   = col & 1023;
          int h   = c >> 6, d = c & 63;
          int b   = row >> 11, t = row & 2047;
          size_t bh = (size_t)(b * 16 + h);
          bf16_t hv = (bf16_t)val;
          if      (sec == 0) Qb[(bh * 2048 + t) * 64 + d] = hv;
          else if (sec == 1) Kb[(bh * 2048 + t) * 64 + d] = hv;
          else               Vt[(bh * 64 + d) * 2048 + t] = hv;
        } else {
          Out[(size_t)row * N + col] = val + bias[col];
        }
      }
    }
  }
}

// ---------------------------------------------------------------- flash attention
// grid = (B*H, T/128), block = 256 (8 waves); each wave owns 16 queries.
__global__ __launch_bounds__(256)
void flash_attn_wmma(const bf16_t* __restrict__ Qb, const bf16_t* __restrict__ Kb,
                     const bf16_t* __restrict__ Vt, bf16_t* __restrict__ Ao) {
  __shared__ __align__(16) bf16_t Pbuf[8][16][LPAD];  // per-wave 16x32 P tile

  const int tid  = threadIdx.x;
  const int lane = tid & 31;
  const int wv   = tid >> 5;
  const int hlf  = lane >> 4;
  const int l16  = lane & 15;
  const int kgA  = hlf * 16;
  const int kgB  = hlf * 16;
  const int bh   = blockIdx.x;                 // b*16 + h
  const int q0   = blockIdx.y * 128 + wv * 16; // this wave's query base

  // Q fragments (A-layout), fixed for whole kernel
  FragU qf[2];
  {
    const bf16_t* qrow = Qb + ((size_t)bh * 2048 + q0 + l16) * 64;
    #pragma unroll
    for (int kc = 0; kc < 2; ++kc) {
      qf[kc].h[0] = *(const v8bf*)(qrow + kc * 32 + kgA);
      qf[kc].h[1] = *(const v8bf*)(qrow + kc * 32 + kgA + 8);
    }
  }

  const v8f zero8 = {0.f,0.f,0.f,0.f,0.f,0.f,0.f,0.f};
  v8f accv[4];                 // O tile 16x64 (4 N-subtiles)
  float mrun[8], lrun[8];
  #pragma unroll
  for (int j = 0; j < 4; ++j) accv[j] = zero8;
  #pragma unroll
  for (int r = 0; r < 8; ++r) { mrun[r] = -3.0e38f; lrun[r] = 0.f; }

  const int qhi = q0 + 15;
  for (int key0 = 0; key0 <= qhi; key0 += 32) {
    // prefetch next key block of K into cache (global_prefetch_b8 path)
    if (key0 + 32 <= qhi) {
      __builtin_prefetch(Kb + ((size_t)bh * 2048 + key0 + 32 + l16) * 64, 0, 0);
      __builtin_prefetch(Kb + ((size_t)bh * 2048 + key0 + 48 + l16) * 64, 0, 0);
    }

    // ---- S = Q * K^T (16 queries x 32 keys), two 16x16 tiles
    AccU s[2];
    #pragma unroll
    for (int ns = 0; ns < 2; ++ns) {
      v8f sv = zero8;
      const bf16_t* krow = Kb + ((size_t)bh * 2048 + key0 + ns * 16 + l16) * 64;
      #pragma unroll
      for (int kc = 0; kc < 2; ++kc) {
        FragU kf;
        kf.h[0] = *(const v8bf*)(krow + kc * 32 + kgB);
        kf.h[1] = *(const v8bf*)(krow + kc * 32 + kgB + 8);
        sv = __builtin_amdgcn_wmma_f32_16x16x32_bf16(
            false, qf[kc].v, false, kf.v, (short)0, sv, false, false);
      }
      s[ns].v = sv;
    }
    // ---- causal mask + scale (1/sqrt(64) = 0.125)
    #pragma unroll
    for (int ns = 0; ns < 2; ++ns) {
      int key = key0 + ns * 16 + l16;
      #pragma unroll
      for (int r = 0; r < 8; ++r) {
        int q = q0 + r + hlf * 8;
        s[ns].f[r] = (key <= q) ? s[ns].f[r] * 0.125f : -3.0e38f;
      }
    }
    // ---- online softmax: row max across 16 lanes (halves hold distinct rows)
    float mt[8];
    #pragma unroll
    for (int r = 0; r < 8; ++r) mt[r] = fmaxf(s[0].f[r], s[1].f[r]);
    #pragma unroll
    for (int m = 8; m >= 1; m >>= 1)
      #pragma unroll
      for (int r = 0; r < 8; ++r) mt[r] = fmaxf(mt[r], __shfl_xor(mt[r], m, 32));

    float alpha[8];
    #pragma unroll
    for (int r = 0; r < 8; ++r) {
      float mn = fmaxf(mrun[r], mt[r]);
      alpha[r] = __expf(mrun[r] - mn);
      mrun[r]  = mn;
    }
    float ps[2][8];
    #pragma unroll
    for (int ns = 0; ns < 2; ++ns)
      #pragma unroll
      for (int r = 0; r < 8; ++r) ps[ns][r] = __expf(s[ns].f[r] - mrun[r]);

    float rs[8];
    #pragma unroll
    for (int r = 0; r < 8; ++r) rs[r] = ps[0][r] + ps[1][r];
    #pragma unroll
    for (int m = 8; m >= 1; m >>= 1)
      #pragma unroll
      for (int r = 0; r < 8; ++r) rs[r] += __shfl_xor(rs[r], m, 32);
    #pragma unroll
    for (int r = 0; r < 8; ++r) lrun[r] = lrun[r] * alpha[r] + rs[r];

    // rescale running O
    #pragma unroll
    for (int j = 0; j < 4; ++j)
      #pragma unroll
      for (int r = 0; r < 8; ++r) accv[j][r] *= alpha[r];

    // ---- transpose P (C-layout -> A-layout) through per-wave LDS
    #pragma unroll
    for (int ns = 0; ns < 2; ++ns)
      #pragma unroll
      for (int r = 0; r < 8; ++r)
        Pbuf[wv][r + hlf * 8][ns * 16 + l16] = (bf16_t)ps[ns][r];
    asm volatile("s_wait_dscnt 0x0" ::: "memory");  // DS in-order per wave; order the cross-lane read
    FragU pf;
    pf.h[0] = *(const v8bf*)&Pbuf[wv][l16][kgA];
    pf.h[1] = *(const v8bf*)&Pbuf[wv][l16][kgA + 8];

    // ---- O += P * V  (V stored transposed [B,H,64,T] => contiguous B-fragments)
    const bf16_t* vbase = Vt + (size_t)bh * 64 * 2048 + key0;
    #pragma unroll
    for (int j = 0; j < 4; ++j) {
      int d = j * 16 + l16;
      FragU vf;
      vf.h[0] = *(const v8bf*)(vbase + (size_t)d * 2048 + kgB);
      vf.h[1] = *(const v8bf*)(vbase + (size_t)d * 2048 + kgB + 8);
      accv[j] = __builtin_amdgcn_wmma_f32_16x16x32_bf16(
          false, pf.v, false, vf.v, (short)0, accv[j], false, false);
    }
  }

  // ---- normalize and store to attn-out [B,T,H,64] == [8192,1024] bf16
  float inv[8];
  #pragma unroll
  for (int r = 0; r < 8; ++r) inv[r] = 1.0f / lrun[r];
  const int b = bh >> 4, h = bh & 15;
  #pragma unroll
  for (int j = 0; j < 4; ++j) {
    #pragma unroll
    for (int r = 0; r < 8; ++r) {
      int q = q0 + r + hlf * 8;
      int d = j * 16 + l16;
      Ao[(((size_t)b * 2048 + q) * 16 + h) * 64 + d] = (bf16_t)(accv[j][r] * inv[r]);
    }
  }
}

// ---------------------------------------------------------------- launch
extern "C" void kernel_launch(void* const* d_in, const int* in_sizes, int n_in,
                              void* d_out, int out_size, void* d_ws, size_t ws_size,
                              hipStream_t stream) {
  (void)in_sizes; (void)n_in; (void)out_size; (void)ws_size;
  const float* x     = (const float*)d_in[0];
  const float* w_qkv = (const float*)d_in[1];
  const float* w_fc  = (const float*)d_in[2];
  const float* b_fc  = (const float*)d_in[3];
  float* out = (float*)d_out;

  size_t off = 0;
  auto alloc = [&](size_t bytes) {
    char* p = (char*)d_ws + off;
    off += (bytes + 255) & ~(size_t)255;
    return p;
  };
  bf16_t* xb  = (bf16_t*)alloc((size_t)8192 * 1024 * 2);  // x bf16
  bf16_t* wqb = (bf16_t*)alloc((size_t)1024 * 3072 * 2);  // w_qkv bf16
  bf16_t* wfb = (bf16_t*)alloc((size_t)1024 * 1024 * 2);  // w_fc bf16
  bf16_t* Qb  = (bf16_t*)alloc((size_t)64 * 2048 * 64 * 2);
  bf16_t* Kb  = (bf16_t*)alloc((size_t)64 * 2048 * 64 * 2);
  bf16_t* Vt  = (bf16_t*)alloc((size_t)64 * 64 * 2048 * 2);
  bf16_t* Ao  = (bf16_t*)alloc((size_t)8192 * 1024 * 2);

  cvt_f32_to_bf16<<<(8192 * 1024 + 255) / 256, 256, 0, stream>>>(x, xb, 8192 * 1024);
  cvt_f32_to_bf16<<<(1024 * 3072 + 255) / 256, 256, 0, stream>>>(w_qkv, wqb, 1024 * 3072);
  cvt_f32_to_bf16<<<(1024 * 1024 + 255) / 256, 256, 0, stream>>>(w_fc, wfb, 1024 * 1024);

  // QKV projection: [8192,1024] x [1024,3072]
  gemm_bf16_wmma<<<dim3(3072 / BN, 8192 / BM), 256, 0, stream>>>(
      xb, wqb, 8192, 3072, 1024, 0, Qb, Kb, Vt, nullptr, nullptr);

  // causal flash attention
  flash_attn_wmma<<<dim3(64, 2048 / 128), 256, 0, stream>>>(Qb, Kb, Vt, Ao);

  // output projection + bias: [8192,1024] x [1024,1024] -> f32 out
  gemm_bf16_wmma<<<dim3(1024 / BN, 8192 / BM), 256, 0, stream>>>(
      Ao, wfb, 8192, 1024, 1024, 1, nullptr, nullptr, nullptr, b_fc, out);
}
